// RNN_36893769073187
// MI455X (gfx1250) — matrix operand order repaired and checked by reading, not compiled, not run on hardware
//
#include <hip/hip_runtime.h>
#include <math.h>

// ---------------------------------------------------------------------------
// RNN: out[:,t,:] = h_t = tanh(x[:,t,:]@Wx + b + h_{t-1}@Wh)
// N=128, T=512, D=1024, H=1024.
//   pass 0: x (fp32)  -> x_bf16   (bandwidth-bound, one shot)
//           Wx,Wh     -> transposed bf16 (so B-fragments load contiguously)
//   pass 1: xW = x@Wx + b as one (65536 x 1024 x 1024) bf16-WMMA GEMM, fp32 acc
//   pass 2: 512 sequential step kernels, each a (128 x 1024 x 1024) bf16-WMMA
//           GEMM with accumulator seeded from xW, then v_tanh_f32.
// ---------------------------------------------------------------------------

#define D_DIM 1024
#define H_DIM 1024
#define T_DIM 512
#define N_BATCH 128
#define M_TOTAL (N_BATCH * T_DIM)   // 65536

typedef __attribute__((ext_vector_type(16))) __bf16 v16bf;
typedef __attribute__((ext_vector_type(8)))  float  v8f;
typedef __attribute__((ext_vector_type(4)))  float  v4f;
typedef __attribute__((ext_vector_type(8)))  unsigned short u16x8; // 16 bytes

union BFrag {            // reinterpret 16 packed bf16 (as ushort) as WMMA operand
    v16bf v;
    u16x8 h[2];
    unsigned short s[16];
};

// float -> bf16 (round-to-nearest-even), pure bit manipulation
__device__ __forceinline__ unsigned short f2bf(float f) {
    unsigned int u = __builtin_bit_cast(unsigned int, f);
    unsigned int lsb = (u >> 16) & 1u;
    u += 0x7fffu + lsb;
    return (unsigned short)(u >> 16);
}

// gfx1250 has a hardware TANH trans op; use it when the builtin exists.
__device__ __forceinline__ float fast_tanh(float x) {
#if __has_builtin(__builtin_amdgcn_tanhf)
    return __builtin_amdgcn_tanhf(x);
#elif __has_builtin(__builtin_amdgcn_tanh_f32)
    return __builtin_amdgcn_tanh_f32(x);
#else
    return tanhf(x);
#endif
}

// ---------------------------------------------------------------------------
// Bulk fp32 -> bf16, 8 elements per thread (2x b128 in, 1x b128 out)
// ---------------------------------------------------------------------------
__global__ __launch_bounds__(256) void cvt_bf16_bulk_kernel(
        const float* __restrict__ src, unsigned short* __restrict__ dst,
        size_t n8) {  // number of 8-element groups
    size_t g = (size_t)blockIdx.x * blockDim.x + threadIdx.x;
    if (g >= n8) return;
    const float* p = src + g * 8;
    v4f f0 = *(const v4f*)(p);
    v4f f1 = *(const v4f*)(p + 4);
    u16x8 o;
#pragma unroll
    for (int i = 0; i < 4; ++i) { o[i] = f2bf(f0[i]); o[4 + i] = f2bf(f1[i]); }
    *(u16x8*)(dst + g * 8) = o;
}

// ---------------------------------------------------------------------------
// Transpose + convert: src (rows x cols) fp32 row-major -> dst (cols x rows) bf16
// ---------------------------------------------------------------------------
__global__ __launch_bounds__(256) void transpose_cvt_kernel(
        const float* __restrict__ src, unsigned short* __restrict__ dst,
        int rows, int cols) {
    size_t idx = (size_t)blockIdx.x * blockDim.x + threadIdx.x;
    size_t total = (size_t)rows * cols;
    if (idx >= total) return;
    int r = (int)(idx / cols);
    int c = (int)(idx % cols);
    dst[(size_t)c * rows + r] = f2bf(src[idx]);
}

// ---------------------------------------------------------------------------
// Pass 1: xW = X @ Wx + b.
// Xb:  (65536, 1024) bf16 row-major.
// WxT: (H=1024 rows, D=1024 cols) bf16 = Wx transposed.
// Block = 8 waves (256 thr). Wave tile = 32x64 (2 M-tiles x 4 N-tiles).
// Waves arranged 4(M) x 2(N): block tile 128x128.
// A-fragment (16-bit A 16x32 layout): row = lane%16, K chunks of 8 at
//   koff=(lane>=16)*8 and koff+16.  B-fragment (32x16): lane n%16 holds
//   column n, K chunks at the same koff, koff+16 as A.
// ---------------------------------------------------------------------------
__global__ __launch_bounds__(256) void gemm_xw_kernel(
        const unsigned short* __restrict__ Xb,
        const unsigned short* __restrict__ WxT,
        const float* __restrict__ bias,
        float* __restrict__ xW) {
    const int lane = threadIdx.x & 31;
    const int wid  = threadIdx.x >> 5;
    const int wm = wid >> 1;          // 0..3
    const int wn = wid & 1;           // 0..1
    const int m0 = blockIdx.y * 128 + wm * 32;
    const int n0 = blockIdx.x * 128 + wn * 64;

    const int arow  = lane & 15;
    const int akoff = (lane >> 4) << 3;   // A: K chunks at akoff, akoff+16
    const int bkoff = (lane >> 4) << 3;   // B: K chunks at bkoff, bkoff+16

    v8f acc[2][4];
#pragma unroll
    for (int i = 0; i < 2; ++i)
#pragma unroll
        for (int j = 0; j < 4; ++j)
            acc[i][j] = (v8f){0.f, 0.f, 0.f, 0.f, 0.f, 0.f, 0.f, 0.f};

    for (int k = 0; k < D_DIM; k += 32) {
        v16bf afrag[2];
#pragma unroll
        for (int i = 0; i < 2; ++i) {
            const unsigned short* pa =
                Xb + (size_t)(m0 + i * 16 + arow) * D_DIM + k + akoff;
            BFrag ua;
            ua.h[0] = *(const u16x8*)(pa);
            ua.h[1] = *(const u16x8*)(pa + 16);
            afrag[i] = ua.v;
        }
        v16bf bfrag[4];
#pragma unroll
        for (int j = 0; j < 4; ++j) {
            const unsigned short* pb =
                WxT + (size_t)(n0 + j * 16 + (lane & 15)) * D_DIM + k + bkoff;
            BFrag ub;
            ub.h[0] = *(const u16x8*)(pb);
            ub.h[1] = *(const u16x8*)(pb + 16);
            bfrag[j] = ub.v;
        }
#pragma unroll
        for (int i = 0; i < 2; ++i)
#pragma unroll
            for (int j = 0; j < 4; ++j)
                acc[i][j] = __builtin_amdgcn_wmma_f32_16x16x32_bf16(
                    false, afrag[i], false, bfrag[j],
                    (short)0, acc[i][j], false, false);
    }

    // Epilogue: D layout -> row-major store, add bias (bias depends on N only)
    const int drow = (lane >> 4) << 3;  // +0 or +8 to M
    const int dcol = lane & 15;
#pragma unroll
    for (int i = 0; i < 2; ++i) {
#pragma unroll
        for (int j = 0; j < 4; ++j) {
            float bv = bias[n0 + j * 16 + dcol];
            float* po = xW + (size_t)(m0 + i * 16 + drow) * H_DIM + n0 + j * 16 + dcol;
#pragma unroll
            for (int r = 0; r < 8; ++r)
                po[(size_t)r * H_DIM] = acc[i][j][r] + bv;
        }
    }
}

// ---------------------------------------------------------------------------
// Pass 2: one timestep. h_next = tanh(xW_t + h_prev @ Wh)
// 512 16x16 output tiles (8 M x 64 N); one tile per wave; 64 blocks x 8 waves.
// Accumulator seeded from xW tile, 32 chained K=32 WMMAs.
// xw_t / out_t point at column t of the (N, T, H) tensors: row stride = T*H.
// ---------------------------------------------------------------------------
__global__ __launch_bounds__(256) void rnn_step_kernel(
        const unsigned short* __restrict__ hprev,  // (128, 1024) bf16
        const unsigned short* __restrict__ WhT,    // (1024, 1024) bf16 (Wh^T)
        const float* __restrict__ xw_t,
        float* __restrict__ out_t,
        unsigned short* __restrict__ hnext) {      // (128, 1024) bf16
    const int lane = threadIdx.x & 31;
    const int wid  = threadIdx.x >> 5;
    const int tile = blockIdx.x * 8 + wid;  // 0..511
    const int m0 = (tile & 7) * 16;
    const int n0 = (tile >> 3) * 16;
    const size_t ldc = (size_t)T_DIM * H_DIM;

    const int drow = (lane >> 4) << 3;
    const int dcol = lane & 15;

    // seed accumulator with the precomputed input projection
    v8f acc;
    {
        const float* pc = xw_t + (size_t)(m0 + drow) * ldc + n0 + dcol;
#pragma unroll
        for (int r = 0; r < 8; ++r) acc[r] = pc[(size_t)r * ldc];
    }

    const int arow  = lane & 15;
    const int akoff = (lane >> 4) << 3;
    const int bkoff = (lane >> 4) << 3;

    for (int k = 0; k < H_DIM; k += 32) {
        BFrag ua, ub;
        const unsigned short* pa = hprev + (size_t)(m0 + arow) * H_DIM + k + akoff;
        ua.h[0] = *(const u16x8*)(pa);
        ua.h[1] = *(const u16x8*)(pa + 16);
        const unsigned short* pb = WhT + (size_t)(n0 + arow) * H_DIM + k + bkoff;
        ub.h[0] = *(const u16x8*)(pb);
        ub.h[1] = *(const u16x8*)(pb + 16);
        acc = __builtin_amdgcn_wmma_f32_16x16x32_bf16(
            false, ua.v, false, ub.v, (short)0, acc, false, false);
    }

    float* po = out_t + (size_t)(m0 + drow) * ldc + n0 + dcol;
    unsigned short* ph = hnext + (size_t)(m0 + drow) * H_DIM + n0 + dcol;
#pragma unroll
    for (int r = 0; r < 8; ++r) {
        float v = fast_tanh(acc[r]);
        po[(size_t)r * ldc] = v;
        ph[(size_t)r * H_DIM] = f2bf(v);
    }
}

// ---------------------------------------------------------------------------
extern "C" void kernel_launch(void* const* d_in, const int* in_sizes, int n_in,
                              void* d_out, int out_size, void* d_ws, size_t ws_size,
                              hipStream_t stream) {
    (void)in_sizes; (void)n_in; (void)out_size; (void)ws_size;
    const float* x  = (const float*)d_in[0];  // (N,T,D)
    const float* h0 = (const float*)d_in[1];  // (N,H)
    const float* Wx = (const float*)d_in[2];  // (D,H)
    const float* Wh = (const float*)d_in[3];  // (H,H)
    const float* b  = (const float*)d_in[4];  // (H,)
    float* out = (float*)d_out;               // (N,T,H)

    char* ws = (char*)d_ws;
    size_t off = 0;
    float* xW = (float*)(ws + off);                    off += (size_t)M_TOTAL * H_DIM * 4;
    unsigned short* Xb  = (unsigned short*)(ws + off); off += (size_t)M_TOTAL * D_DIM * 2;
    unsigned short* WxT = (unsigned short*)(ws + off); off += (size_t)H_DIM * D_DIM * 2;
    unsigned short* WhT = (unsigned short*)(ws + off); off += (size_t)H_DIM * H_DIM * 2;
    unsigned short* hA  = (unsigned short*)(ws + off); off += (size_t)N_BATCH * H_DIM * 2;
    unsigned short* hB  = (unsigned short*)(ws + off);

    // pass 0a: x -> bf16 (64M elements, 8 per thread)
    {
        size_t n8 = (size_t)M_TOTAL * D_DIM / 8;
        cvt_bf16_bulk_kernel<<<(unsigned)((n8 + 255) / 256), 256, 0, stream>>>(x, Xb, n8);
    }
    // pass 0b: h0 -> bf16 (128K elements)
    {
        size_t n8 = (size_t)N_BATCH * H_DIM / 8;
        cvt_bf16_bulk_kernel<<<(unsigned)((n8 + 255) / 256), 256, 0, stream>>>(h0, hA, n8);
    }
    // pass 0c: weight transpose + bf16 conversion
    {
        int total = D_DIM * H_DIM;
        transpose_cvt_kernel<<<(total + 255) / 256, 256, 0, stream>>>(Wx, WxT, D_DIM, H_DIM);
        transpose_cvt_kernel<<<(total + 255) / 256, 256, 0, stream>>>(Wh, WhT, H_DIM, H_DIM);
    }
    // pass 1: big GEMM  xW = x@Wx + b
    {
        dim3 grid(H_DIM / 128, M_TOTAL / 128);  // (8, 512)
        gemm_xw_kernel<<<grid, 256, 0, stream>>>(Xb, WxT, b, xW);
    }
    // pass 2: sequential recurrence, ping-pong bf16 hidden state
    unsigned short* hp = hA;
    unsigned short* hn = hB;
    for (int t = 0; t < T_DIM; ++t) {
        rnn_step_kernel<<<64, 256, 0, stream>>>(
            hp, WhT, xW + (size_t)t * H_DIM, out + (size_t)t * H_DIM, hn);
        unsigned short* tmp = hp; hp = hn; hn = tmp;
    }
}
